// MultiHeadGAT_61323543052452
// MI455X (gfx1250) — matrix operand order repaired and checked by reading, not compiled, not run on hardware
//
#include <hip/hip_runtime.h>
#include <math.h>
#include <stdint.h>

// ---------------------------------------------------------------------------
// MultiHeadGAT forward for gfx1250 (MI455X, wave32, WMMA).
// Heavy GEMMs: v_wmma_f32_16x16x32_bf16, 128x64 block tile, 32x32 per wave
// (4 accumulators -> 4 WMMAs per 8 LDS b128 reads), double-buffered LDS with
// async global->LDS staging (ASYNCcnt) when the toolchain exposes it.
// alpha@h is algebraically reduced to colsum(alpha)@h (outputs are pooled).
// Quantile(0.7) via in-LDS bitonic sort per attention row.
// ---------------------------------------------------------------------------

typedef __attribute__((ext_vector_type(16))) __bf16 v16bf;
typedef __attribute__((ext_vector_type(8)))  float  v8f;
typedef __attribute__((ext_vector_type(4)))  int    v4i;

static const int B_  = 16;
static const int L_  = 1024;
static const int D_  = 768;
static const int H_  = 256;
static const int NH_ = 4;
static const int OUT_= 768;

// ---------------------------------------------------------------- async LDS
#if defined(__gfx1250__) && __has_builtin(__builtin_amdgcn_global_load_async_to_lds_b128) && __has_builtin(__builtin_amdgcn_s_wait_asynccnt)
#define HAS_ASYNC_LDS 1
#else
#define HAS_ASYNC_LDS 0
#endif

// Builtin signature (from clang): (v4i addrspace(1)*, v4i addrspace(3)*, Ii, Ii)
typedef __attribute__((address_space(1))) v4i as1_v4i;
typedef __attribute__((address_space(3))) v4i as3_v4i;

// 16-byte global -> LDS copy; async (ASYNCcnt) when available.
__device__ __forceinline__ void copy16_g2l(const void* g, void* l) {
#if HAS_ASYNC_LDS
  __builtin_amdgcn_global_load_async_to_lds_b128(
      (as1_v4i*)(uintptr_t)g,
      (as3_v4i*)(uint32_t)(uintptr_t)l,   // generic LDS addr: offset in [31:0]
      0, 0);
#else
  *(uint4*)l = *(const uint4*)g;
#endif
}
__device__ __forceinline__ void g2l_wait() {
#if HAS_ASYNC_LDS
  __builtin_amdgcn_s_wait_asynccnt(0);
#endif
}

// ---------------------------------------------------------------- converts
__global__ __launch_bounds__(256) void k_f32_to_bf16(
    const float* __restrict__ src, __bf16* __restrict__ dst, int n) {
  int i = blockIdx.x * 256 + threadIdx.x;
  if (i < n) dst[i] = (__bf16)src[i];
}

__global__ __launch_bounds__(256) void k_zero(float* __restrict__ p, int n) {
  int i = blockIdx.x * 256 + threadIdx.x;
  if (i < n) p[i] = 0.0f;
}

// ---------------------------------------------------------------- WMMA GEMM
// C[M,N] = A[M,K] * op(B) (+bias), bf16 inputs, f32 accumulate.
// BT==0: B is K x N row-major.  BT==1: B is N x K row-major (A*B^T).
// EPI==0: store f32.  EPI==1: store bf16.
// Block tile 128x64, 8 waves as 4(M) x 2(N), each wave 32x32 (4 accums).
template<int BT, int EPI>
__global__ __launch_bounds__(256) void k_gemm_wmma(
    const __bf16* __restrict__ Ag, const __bf16* __restrict__ Bg,
    void* __restrict__ Cg, const float* __restrict__ bias,
    int M, int N, int K, long sA, long sB, long sC)
{
  __shared__ __bf16 As[2][128][40];  // row stride 40 (80B, 16B aligned)
  __shared__ __bf16 Bs[2][64][40];   // stored as N x K ("B transposed")

  const int tid  = threadIdx.x;
  const int lane = tid & 31;
  const int wave = tid >> 5;
  const int m0 = blockIdx.y * 128;
  const int n0 = blockIdx.x * 64;
  const long bz = blockIdx.z;
  Ag += bz * sA;
  Bg += bz * sB;

  const int waveM = wave >> 1;   // 0..3
  const int waveN = wave & 1;    // 0..1

  v8f acc00 = {}, acc01 = {}, acc10 = {}, acc11 = {};
  const int numK = K >> 5;

  auto stageA = [&](int kt, int buf) {
    // 128x32 bf16 = 512 uint4 slots; 2 per thread.
#pragma unroll
    for (int r = 0; r < 2; ++r) {
      const int s  = tid + 256 * r;
      const int row = s >> 2;
      const int cg  = (s & 3) << 3;
      copy16_g2l(Ag + (long)(m0 + row) * K + (kt << 5) + cg, &As[buf][row][cg]);
    }
  };
  auto stageB = [&](int kt, int buf) {
    if (BT) {            // B stored N x K: rows load straight into Bs[n][k]
      const int row = tid >> 2;
      const int cg  = (tid & 3) << 3;
      copy16_g2l(Bg + (long)(n0 + row) * K + (kt << 5) + cg, &Bs[buf][row][cg]);
    } else {             // B stored K x N: transpose while staging (scalar)
      const int k  = tid >> 3;           // 32 k rows
      const int ng = (tid & 7) << 3;     // 8 groups of 8 columns
      union { uint4 u; __bf16 h[8]; } t;
      t.u = *(const uint4*)(Bg + (long)((kt << 5) + k) * N + n0 + ng);
#pragma unroll
      for (int q = 0; q < 8; ++q) Bs[buf][ng + q][k] = t.h[q];
    }
  };

  stageA(0, 0); stageB(0, 0);
  g2l_wait();
  __syncthreads();

  for (int kt = 0; kt < numK; ++kt) {
    const int buf = kt & 1;
    if (kt + 1 < numK) { stageA(kt + 1, buf ^ 1); stageB(kt + 1, buf ^ 1); }

    // A fragment (16-bit 16x32): lane l -> row l%16; elems 0..7 = K[kb..+7],
    // elems 8..15 = K[16+kb..], kb = 8*(l/16). Two 16B LDS reads each.
    union { v16bf v; uint4 u[2]; } a0, a1, b0, b1;
    {
      const int m  = lane & 15;
      const int kb = (lane >> 4) << 3;
      const int ra = waveM * 32;
      a0.u[0] = *(const uint4*)(&As[buf][ra +      m][kb]);
      a0.u[1] = *(const uint4*)(&As[buf][ra +      m][16 + kb]);
      a1.u[0] = *(const uint4*)(&As[buf][ra + 16 + m][kb]);
      a1.u[1] = *(const uint4*)(&As[buf][ra + 16 + m][16 + kb]);
      // B fragment (32x16): lane l -> col l%16, same K order as A:
      // elems 0..7 = K[kb..+7], elems 8..15 = K[16+kb..]. Two 16B reads.
      const int nb = waveN * 32 + (lane & 15);
      const int ks = (lane >> 4) << 3;
      b0.u[0] = *(const uint4*)(&Bs[buf][nb][ks]);
      b0.u[1] = *(const uint4*)(&Bs[buf][nb][ks + 16]);
      b1.u[0] = *(const uint4*)(&Bs[buf][nb + 16][ks]);
      b1.u[1] = *(const uint4*)(&Bs[buf][nb + 16][ks + 16]);
    }
    acc00 = __builtin_amdgcn_wmma_f32_16x16x32_bf16(false, a0.v, false, b0.v, (short)0, acc00, false, false);
    acc01 = __builtin_amdgcn_wmma_f32_16x16x32_bf16(false, a0.v, false, b1.v, (short)0, acc01, false, false);
    acc10 = __builtin_amdgcn_wmma_f32_16x16x32_bf16(false, a1.v, false, b0.v, (short)0, acc10, false, false);
    acc11 = __builtin_amdgcn_wmma_f32_16x16x32_bf16(false, a1.v, false, b1.v, (short)0, acc11, false, false);

    g2l_wait();
    __syncthreads();
  }

  // Epilogue. C/D layout: VGPR v -> M = v + 8*(lane/16); N = lane%16.
  const int col0  = n0 + waveN * 32 + (lane & 15);
  const int rbase = m0 + waveM * 32 + ((lane >> 4) << 3);
  const float bv0 = bias ? bias[col0] : 0.0f;
  const float bv1 = bias ? bias[col0 + 16] : 0.0f;
  const long cb   = bz * sC;
#pragma unroll
  for (int v = 0; v < 8; ++v) {
    const long r0 = rbase + v;
    const long r1 = rbase + 16 + v;
    const float x00 = acc00[v] + bv0, x01 = acc01[v] + bv1;
    const float x10 = acc10[v] + bv0, x11 = acc11[v] + bv1;
    if (EPI == 0) {
      ((float*)Cg)[cb + r0 * N + col0]      = x00;
      ((float*)Cg)[cb + r0 * N + col0 + 16] = x01;
      ((float*)Cg)[cb + r1 * N + col0]      = x10;
      ((float*)Cg)[cb + r1 * N + col0 + 16] = x11;
    } else {
      ((__bf16*)Cg)[cb + r0 * N + col0]      = (__bf16)x00;
      ((__bf16*)Cg)[cb + r0 * N + col0 + 16] = (__bf16)x01;
      ((__bf16*)Cg)[cb + r1 * N + col0]      = (__bf16)x10;
      ((__bf16*)Cg)[cb + r1 * N + col0 + 16] = (__bf16)x11;
    }
  }
}

// ------------------------------------------------------- LayerNorm (H=256)
// In-place on fp32 rows; also emits bf16 copy. One wave per row, 8 rows/block.
__global__ __launch_bounds__(256) void k_layernorm_rows(
    float* __restrict__ hio, __bf16* __restrict__ hb,
    const float* __restrict__ g, const float* __restrict__ bvec)
{
  const int wave = threadIdx.x >> 5;
  const int lane = threadIdx.x & 31;
  const long row = (long)blockIdx.x * 8 + wave;
  float x[8], s = 0.f, sq = 0.f;
#pragma unroll
  for (int u = 0; u < 8; ++u) {
    x[u] = hio[row * H_ + lane + 32 * u];
    s += x[u]; sq += x[u] * x[u];
  }
#pragma unroll
  for (int m = 16; m > 0; m >>= 1) {
    s  += __shfl_xor(s,  m, 32);
    sq += __shfl_xor(sq, m, 32);
  }
  const float mu  = s * (1.0f / H_);
  const float var = sq * (1.0f / H_) - mu * mu;
  const float inv = rsqrtf(var + 1e-5f);
#pragma unroll
  for (int u = 0; u < 8; ++u) {
    const int c = lane + 32 * u;
    const float y = (x[u] - mu) * inv * g[c] + bvec[c];
    hio[row * H_ + c] = y;
    hb [row * H_ + c] = (__bf16)y;
  }
}

// ----------------------------------------- e_src / e_dst GEMVs (one wave/row)
__global__ __launch_bounds__(256) void k_edge_scores(
    const float* __restrict__ h, const float* __restrict__ Watt,
    float* __restrict__ esrc, float* __restrict__ edst)
{
  const int wave = threadIdx.x >> 5;
  const int lane = threadIdx.x & 31;
  const long row = (long)blockIdx.x * 8 + wave;
  float s1 = 0.f, s2 = 0.f;
#pragma unroll
  for (int u = 0; u < 8; ++u) {
    const int c = lane + 32 * u;
    const float hv = h[row * H_ + c];
    s1 += hv * Watt[c];
    s2 += hv * Watt[H_ + c];
  }
#pragma unroll
  for (int m = 16; m > 0; m >>= 1) {
    s1 += __shfl_xor(s1, m, 32);
    s2 += __shfl_xor(s2, m, 32);
  }
  if (lane == 0) { esrc[row] = s1; edst[row] = s2; }
}

// --------------------------------------------- attention row (one block/row)
// sigmoid -> bitonic-sort quantile(0.7) -> mask -> leaky_relu -> softmax ->
// atomic column-sum of alpha.
__global__ __launch_bounds__(256) void k_attention_row(
    const float* __restrict__ Abuf, const float* __restrict__ esrc,
    const float* __restrict__ edst, const float* __restrict__ battp,
    float* __restrict__ colsum)
{
  __shared__ float sval[L_];
  __shared__ float ssort[L_];
  __shared__ float red[256];
  __shared__ float sdelta;

  const int tid  = threadIdx.x;
  const long row = blockIdx.x;          // b*L + i
  const int  b   = (int)(row >> 10);
  const int  i   = (int)(row & (L_ - 1));
  const float batt = battp[0];
  const float* arow = Abuf + row * (long)L_;

  for (int j = tid; j < L_; j += 256) {
    const float sg = 1.0f / (1.0f + expf(-arow[j]));
    sval[j]  = sg;
    ssort[j] = sg;
  }
  __syncthreads();

  // Bitonic sort (ascending), 1024 elems, 256 threads x 4 lanes each.
  for (int k = 2; k <= L_; k <<= 1) {
    for (int j = k >> 1; j > 0; j >>= 1) {
      for (int t = tid; t < L_; t += 256) {
        const int ixj = t ^ j;
        if (ixj > t) {
          const float a = ssort[t], c = ssort[ixj];
          const bool up = ((t & k) == 0);
          if ((a > c) == up) { ssort[t] = c; ssort[ixj] = a; }
        }
      }
      __syncthreads();
    }
  }
  if (tid == 0) {
    const float pos = 0.7f * (float)(L_ - 1);
    const int lo = (int)pos;
    const float fr = pos - (float)lo;
    sdelta = ssort[lo] + fr * (ssort[lo + 1] - ssort[lo]);
  }
  __syncthreads();
  const float delta = sdelta;
  const float ei = esrc[row];
  __syncthreads();   // done reading ssort as sorted data; reuse below

  float lmax = -3.0e38f;
  for (int j = tid; j < L_; j += 256) {
    float ev;
    if (sval[j] > delta || j == i) {
      const float raw = ei + edst[(long)b * L_ + j] + batt;
      ev = raw > 0.f ? raw : 0.01f * raw;           // leaky_relu(0.01)
    } else {
      ev = -3.0e38f;                                 // acts as -inf
    }
    ssort[j] = ev;
    lmax = fmaxf(lmax, ev);
  }
  red[tid] = lmax; __syncthreads();
  for (int st = 128; st > 0; st >>= 1) {
    if (tid < st) red[tid] = fmaxf(red[tid], red[tid + st]);
    __syncthreads();
  }
  const float mx = red[0];
  __syncthreads();

  float lsum = 0.f;
  for (int j = tid; j < L_; j += 256) {
    const float p = expf(ssort[j] - mx);   // masked -> exp(-huge) == 0
    sval[j] = p;
    lsum += p;
  }
  red[tid] = lsum; __syncthreads();
  for (int st = 128; st > 0; st >>= 1) {
    if (tid < st) red[tid] += red[tid + st];
    __syncthreads();
  }
  const float inv = 1.0f / red[0];
  for (int j = tid; j < L_; j += 256)
    atomicAdd(&colsum[(long)b * L_ + j], sval[j] * inv);
}

// ---------------------------- pooled head output: mean = colsum(alpha)@h / L
__global__ __launch_bounds__(256) void k_colsum_gemv(
    const float* __restrict__ colsum, const float* __restrict__ h,
    float* __restrict__ meanh)
{
  const int b = blockIdx.x;
  const int c = threadIdx.x;   // H_ == 256
  const float* cs = colsum + (long)b * L_;
  const float* hb = h + (long)b * L_ * H_;
  float acc = 0.f;
  for (int l = 0; l < L_; ++l) acc += cs[l] * hb[(long)l * H_ + c];
  meanh[(long)b * H_ + c] = acc * (1.0f / L_);
}

// ------------------------------------------- final FC + LayerNorm (tiny)
__global__ __launch_bounds__(256) void k_final(
    const float* __restrict__ meanh,   // (NH,B,H)
    const float* __restrict__ Wl, const float* __restrict__ bl,
    const float* __restrict__ Ws, const float* __restrict__ bs,
    const float* __restrict__ Wc, const float* __restrict__ bc,
    const float* __restrict__ gl, const float* __restrict__ bbl,
    const float* __restrict__ gs, const float* __restrict__ bbs,
    const float* __restrict__ gc, const float* __restrict__ bbc,
    float* __restrict__ out)
{
  __shared__ float cat[NH_ * H_];
  __shared__ float red[256];
  const int b = blockIdx.x;
  const int tid = threadIdx.x;
  for (int idx = tid; idx < NH_ * H_; idx += 256) {
    const int hd = idx >> 8, c = idx & 255;
    cat[idx] = meanh[((long)hd * B_ + b) * H_ + c];
  }
  __syncthreads();

  for (int which = 0; which < 3; ++which) {
    const float *W, *bias, *g, *beta, *vec; float* op; int klen;
    if (which == 0)      { W=Wl; bias=bl; g=gl; beta=bbl; vec=cat;          klen=2*H_;   op=out; }
    else if (which == 1) { W=Ws; bias=bs; g=gs; beta=bbs; vec=cat + 2*H_;   klen=2*H_;   op=out + (long)B_*OUT_; }
    else                 { W=Wc; bias=bc; g=gc; beta=bbc; vec=cat;          klen=NH_*H_; op=out + 2L*B_*OUT_; }
    float local[3], s = 0.f, sq = 0.f;
#pragma unroll
    for (int r = 0; r < 3; ++r) {
      const int oc = tid + r * 256;
      float acc = bias[oc];
      for (int k = 0; k < klen; ++k) acc += vec[k] * W[(long)k * OUT_ + oc];
      local[r] = acc; s += acc; sq += acc * acc;
    }
    red[tid] = s; __syncthreads();
    for (int st = 128; st > 0; st >>= 1) { if (tid < st) red[tid] += red[tid + st]; __syncthreads(); }
    const float mu = red[0] * (1.0f / OUT_);
    __syncthreads();
    red[tid] = sq; __syncthreads();
    for (int st = 128; st > 0; st >>= 1) { if (tid < st) red[tid] += red[tid + st]; __syncthreads(); }
    const float var = red[0] * (1.0f / OUT_) - mu * mu;
    const float inv = rsqrtf(var + 1e-5f);
    __syncthreads();
#pragma unroll
    for (int r = 0; r < 3; ++r) {
      const int oc = tid + r * 256;
      op[(long)b * OUT_ + oc] = (local[r] - mu) * inv * g[oc] + beta[oc];
    }
    __syncthreads();
  }
}

// ---------------------------------------------------------------- host side
extern "C" void kernel_launch(void* const* d_in, const int* in_sizes, int n_in,
                              void* d_out, int out_size, void* d_ws, size_t ws_size,
                              hipStream_t stream) {
  (void)in_sizes; (void)n_in; (void)out_size; (void)ws_size;
  const float* x    = (const float*)d_in[0];
  const float* Wfc  = (const float*)d_in[1];
  const float* bfc  = (const float*)d_in[2];
  const float* lng  = (const float*)d_in[3];
  const float* lnb  = (const float*)d_in[4];
  const float* adjw = (const float*)d_in[5];
  const float* Watt = (const float*)d_in[6];
  const float* batt = (const float*)d_in[7];
  const float* fcLW = (const float*)d_in[8];
  const float* fcLb = (const float*)d_in[9];
  const float* fcSW = (const float*)d_in[10];
  const float* fcSb = (const float*)d_in[11];
  const float* fcCW = (const float*)d_in[12];
  const float* fcCb = (const float*)d_in[13];
  const float* nLg  = (const float*)d_in[14];
  const float* nLb  = (const float*)d_in[15];
  const float* nSg  = (const float*)d_in[16];
  const float* nSb  = (const float*)d_in[17];
  const float* nCg  = (const float*)d_in[18];
  const float* nCb  = (const float*)d_in[19];

  char* ws = (char*)d_ws;
  size_t off = 0;
  auto mk = [&](size_t bytes) -> void* {
    void* p = (void*)(ws + off);
    off = (off + bytes + 255) & ~(size_t)255;
    return p;
  };
  __bf16* xb    = (__bf16*)mk((size_t)B_ * L_ * D_ * 2);
  __bf16* wfcb  = (__bf16*)mk((size_t)NH_ * D_ * H_ * 2);
  __bf16* adjb  = (__bf16*)mk((size_t)NH_ * H_ * H_ * 2);
  float*  hpre  = (float*) mk((size_t)B_ * L_ * H_ * 4);   // h (post-LN, in-place)
  __bf16* hb    = (__bf16*)mk((size_t)B_ * L_ * H_ * 2);
  __bf16* tb    = (__bf16*)mk((size_t)B_ * L_ * H_ * 2);
  float*  Abuf  = (float*) mk((size_t)B_ * L_ * L_ * 4);
  float*  esrc  = (float*) mk((size_t)B_ * L_ * 4);
  float*  edst  = (float*) mk((size_t)B_ * L_ * 4);
  float*  csum  = (float*) mk((size_t)B_ * L_ * 4);
  float*  meanh = (float*) mk((size_t)NH_ * B_ * H_ * 4);

  const int nx = B_ * L_ * D_;
  const int nw = NH_ * D_ * H_;
  const int na = NH_ * H_ * H_;
  k_f32_to_bf16<<<(nx + 255) / 256, 256, 0, stream>>>(x, xb, nx);
  k_f32_to_bf16<<<(nw + 255) / 256, 256, 0, stream>>>(Wfc, wfcb, nw);
  k_f32_to_bf16<<<(na + 255) / 256, 256, 0, stream>>>(adjw, adjb, na);

  const int ML = B_ * L_;                  // 16384 rows
  const dim3 gFC(H_ / 64, ML / 128, 1);    // (4, 128)
  const dim3 gA (L_ / 64, L_ / 128, B_);   // (16, 8, 16)

  for (int hd = 0; hd < NH_; ++hd) {
    // h_pre = x @ Wfc[hd] + bfc[hd]   (bf16 WMMA, f32 out)
    k_gemm_wmma<0, 0><<<gFC, 256, 0, stream>>>(
        xb, wfcb + (long)hd * D_ * H_, hpre, bfc + hd * H_,
        ML, H_, D_, 0, 0, 0);
    // h = LN(h_pre) in-place (+ bf16 copy)
    k_layernorm_rows<<<ML / 8, 256, 0, stream>>>(hpre, hb, lng + hd * H_, lnb + hd * H_);
    // t = h @ adjw[hd]  (bf16 out)
    k_gemm_wmma<0, 1><<<gFC, 256, 0, stream>>>(
        hb, adjb + (long)hd * H_ * H_, tb, nullptr,
        ML, H_, H_, 0, 0, 0);
    // e_src/e_dst GEMVs
    k_edge_scores<<<ML / 8, 256, 0, stream>>>(hpre, Watt + hd * 2 * H_, esrc, edst);
    // A = t @ h^T per batch  (NT, f32 out, fully async-staged)
    k_gemm_wmma<1, 0><<<gA, 256, 0, stream>>>(
        tb, hb, Abuf, nullptr,
        L_, L_, H_, (long)L_ * H_, (long)L_ * H_, (long)L_ * L_);
    // colsum(alpha)
    k_zero<<<ML / 256, 256, 0, stream>>>(csum, ML);
    k_attention_row<<<ML, 256, 0, stream>>>(Abuf, esrc, edst, batt + hd, csum);
    // pooled head output
    k_colsum_gemv<<<B_, 256, 0, stream>>>(csum, hpre, meanh + (long)hd * B_ * H_);
  }

  k_final<<<B_, 256, 0, stream>>>(meanh,
      fcLW, fcLb, fcSW, fcSb, fcCW, fcCb,
      nLg, nLb, nSg, nSb, nCg, nCb, (float*)d_out);
}
